// GCN_64768106823755
// MI455X (gfx1250) — matrix operand order repaired and checked by reading, not compiled, not run on hardware
//
#include <hip/hip_runtime.h>
#include <hip/hip_bf16.h>

#define N_NODES   50000
#define N_EDGES   600000
#define F_IN      128
#define H_DIM     256
#define N_CLASSES 16
#define N_GRAPHS  64

typedef __attribute__((ext_vector_type(16))) __bf16 v16bf;
typedef __attribute__((ext_vector_type(8)))  __bf16 v8bf;
typedef __attribute__((ext_vector_type(8)))  float  v8f;

__device__ __forceinline__ __bf16 f2bf(float f) {
    union { float f; unsigned int u; } v; v.f = f;
    union { unsigned short s; __bf16 b; } o;
    o.s = (unsigned short)(v.u >> 16);   // truncate-to-bf16
    return o.b;
}

// ---------------------------------------------------------------------------
// Re-tile fp32 weights [K,N] -> bf16 WMMA-B layout [K/32][N][32] so that one
// lane's B-fragment (16 consecutive k at fixed n) is 32 contiguous bytes.
// ---------------------------------------------------------------------------
__global__ void convert_w(const float* __restrict__ W, __bf16* __restrict__ out,
                          int K, int N) {
    int i = blockIdx.x * blockDim.x + threadIdx.x;
    if (i >= K * N) return;
    int k = i / N, n = i - k * N;
    out[(size_t)(k >> 5) * (N * 32) + n * 32 + (k & 31)] = f2bf(W[i]);
}

// ---------------------------------------------------------------------------
// Scatter-mean layer 1: one wave per edge, 128 features (4 per lane).
// ---------------------------------------------------------------------------
__global__ void scatter_mean1(const float* __restrict__ x,
                              const int*   __restrict__ ei,
                              float* __restrict__ msg1,
                              float* __restrict__ cnt) {
    int e = blockIdx.x * (blockDim.x >> 5) + (threadIdx.x >> 5);
    if (e >= N_EDGES) return;
    int lane = threadIdx.x & 31;
    int src = ei[e];
    int dst = ei[N_EDGES + e];
    const float4* xs = (const float4*)(x + (size_t)src * F_IN);
    float4 v = xs[lane];
    float* mrow = msg1 + (size_t)dst * F_IN + lane * 4;
    atomicAdd(mrow + 0, v.x);
    atomicAdd(mrow + 1, v.y);
    atomicAdd(mrow + 2, v.z);
    atomicAdd(mrow + 3, v.w);
    if (lane == 0) atomicAdd(cnt + dst, 1.0f);
}

// ---------------------------------------------------------------------------
// Scatter-mean layer 2: one wave per edge, 256 features (8 per lane).
// ---------------------------------------------------------------------------
__global__ void scatter_mean2(const float* __restrict__ h1,
                              const int*   __restrict__ ei,
                              float* __restrict__ msg2) {
    int e = blockIdx.x * (blockDim.x >> 5) + (threadIdx.x >> 5);
    if (e >= N_EDGES) return;
    int lane = threadIdx.x & 31;
    int src = ei[e];
    int dst = ei[N_EDGES + e];
    const float4* hs = (const float4*)(h1 + (size_t)src * H_DIM);
    float4 v0 = hs[2 * lane + 0];
    float4 v1 = hs[2 * lane + 1];
    float* mrow = msg2 + (size_t)dst * H_DIM + lane * 8;
    atomicAdd(mrow + 0, v0.x); atomicAdd(mrow + 1, v0.y);
    atomicAdd(mrow + 2, v0.z); atomicAdd(mrow + 3, v0.w);
    atomicAdd(mrow + 4, v1.x); atomicAdd(mrow + 5, v1.y);
    atomicAdd(mrow + 6, v1.z); atomicAdd(mrow + 7, v1.w);
}

// ---------------------------------------------------------------------------
// GEMM layer 1:  h1 = relu(mean1 @ W1_l + x @ W1_r + b1)
// Block = 512 thr = 16 waves = the 16 column tiles of one 16-row M tile.
// All waves share the same 16 A rows -> stage them once as bf16 in LDS
// (mean pre-scaled by 1/cnt during staging). 50000 % 16 == 0 -> no guards,
// EXEC all-ones for every WMMA.
// ---------------------------------------------------------------------------
__global__ void __launch_bounds__(512)
gemm1_wmma(const float* __restrict__ msg1,
           const float* __restrict__ cnt,
           const float* __restrict__ x,
           const __bf16* __restrict__ Wl_t,  // tiled [F_IN/32][H_DIM][32]
           const __bf16* __restrict__ Wr_t,
           const float* __restrict__ b,
           float* __restrict__ h1) {
    __shared__ __bf16 sA[2][16][F_IN];       // 8 KB: [0]=mean(scaled), [1]=x
    const int tid = threadIdx.x;
    const int m0  = blockIdx.x * 16;

    // Cooperative stage: 2048 elements per matrix, 512 threads.
    for (int i = tid; i < 16 * F_IN; i += 512) {
        int r = i >> 7, k = i & (F_IN - 1);
        float invr = 1.0f / fmaxf(cnt[m0 + r], 1.0f);
        sA[0][r][k] = f2bf(msg1[(size_t)(m0 + r) * F_IN + k] * invr);
        sA[1][r][k] = f2bf(x[(size_t)(m0 + r) * F_IN + k]);
    }
    __syncthreads();

    const int wave = tid >> 5;
    const int lane = tid & 31;
    const int hf   = lane >> 4;
    const int lrow = lane & 15;
    const int n    = wave * 16 + lrow;

    v8f c;
    float bias = b[n];
    #pragma unroll
    for (int r = 0; r < 8; ++r) c[r] = bias;

    #pragma unroll
    for (int k0 = 0; k0 < F_IN; k0 += 32) {
        // A fragment = two contiguous 8-elt (16 B) chunks from LDS.
        v8bf am0 = *(const v8bf*)&sA[0][lrow][k0 + hf * 16];
        v8bf am1 = *(const v8bf*)&sA[0][lrow][k0 + hf * 16 + 8];
        v8bf ax0 = *(const v8bf*)&sA[1][lrow][k0 + hf * 16];
        v8bf ax1 = *(const v8bf*)&sA[1][lrow][k0 + hf * 16 + 8];
        v16bf a_mean = __builtin_shufflevector(am0, am1,
            0,1,2,3,4,5,6,7,8,9,10,11,12,13,14,15);
        v16bf a_x = __builtin_shufflevector(ax0, ax1,
            0,1,2,3,4,5,6,7,8,9,10,11,12,13,14,15);
        // B fragment = 32 contiguous bytes from the tiled bf16 weights.
        size_t boff = (size_t)(k0 >> 5) * (H_DIM * 32) + n * 32 + hf * 16;
        v16bf bl = *(const v16bf*)&Wl_t[boff];
        v16bf br = *(const v16bf*)&Wr_t[boff];
        c = __builtin_amdgcn_wmma_f32_16x16x32_bf16(false, a_mean, false, bl,
                                                    (short)0, c, false, false);
        c = __builtin_amdgcn_wmma_f32_16x16x32_bf16(false, a_x, false, br,
                                                    (short)0, c, false, false);
    }

    // C/D layout: VGPR r, lanes 0-15 -> M=r, lanes 16-31 -> M=r+8, col=lane&15.
    #pragma unroll
    for (int r = 0; r < 8; ++r) {
        int row = m0 + r + (hf << 3);
        h1[(size_t)row * H_DIM + n] = fmaxf(c[r], 0.0f);
    }
}

// ---------------------------------------------------------------------------
// GEMM layer 2 + fused pooling:
//   h2 = relu(mean2 @ W2_l + h1 @ W2_r + b2);  pool_sum[batch] += h2
// One wave per 16-node tile (single 16-wide column tile, K=256).
// No cross-wave A reuse here, so A converts at load (contiguous b128 runs);
// B comes from the tiled bf16 weights.
// ---------------------------------------------------------------------------
__global__ void __launch_bounds__(256)
gemm2_wmma_pool(const float* __restrict__ msg2,
                const float* __restrict__ cnt,
                const float* __restrict__ h1,
                const __bf16* __restrict__ Wl_t,  // tiled [H_DIM/32][16][32]
                const __bf16* __restrict__ Wr_t,
                const float* __restrict__ b,
                const int*   __restrict__ batch,
                float* __restrict__ pool_sum) {
    const int gwave = blockIdx.x * (blockDim.x >> 5) + (threadIdx.x >> 5);
    if (gwave >= N_NODES / 16) return;           // wave-uniform exit
    const int lane = threadIdx.x & 31;
    const int hf   = lane >> 4;
    const int lrow = lane & 15;
    const int m0   = gwave * 16;
    const int n    = lrow;

    v8f c;
    float bias = b[n];
    #pragma unroll
    for (int r = 0; r < 8; ++r) c[r] = bias;

    const int   arow = m0 + lrow;
    const float inv  = 1.0f / fmaxf(cnt[arow], 1.0f);
    const float* mrowp = msg2 + (size_t)arow * H_DIM;
    const float* hrowp = h1   + (size_t)arow * H_DIM;

    #pragma unroll
    for (int k0 = 0; k0 < H_DIM; k0 += 32) {
        v16bf a_mean, a_h, bl, br;
        // A fragment: elements 0..7 at k0+hf*16+j, 8..15 at k0+hf*16+8+j.
        #pragma unroll
        for (int j = 0; j < 8; ++j) {
            int ka = k0 + (hf << 4) + j;
            int kb = ka + 8;
            a_mean[j]     = f2bf(mrowp[ka] * inv);
            a_mean[8 + j] = f2bf(mrowp[kb] * inv);
            a_h[j]        = f2bf(hrowp[ka]);
            a_h[8 + j]    = f2bf(hrowp[kb]);
        }
        size_t boff = (size_t)(k0 >> 5) * (N_CLASSES * 32) + n * 32 + hf * 16;
        bl = *(const v16bf*)&Wl_t[boff];
        br = *(const v16bf*)&Wr_t[boff];
        c = __builtin_amdgcn_wmma_f32_16x16x32_bf16(false, a_mean, false, bl,
                                                    (short)0, c, false, false);
        c = __builtin_amdgcn_wmma_f32_16x16x32_bf16(false, a_h, false, br,
                                                    (short)0, c, false, false);
    }

    // relu + pool directly (h2 never stored)
    #pragma unroll
    for (int r = 0; r < 8; ++r) {
        int row = m0 + r + (hf << 3);
        float v = fmaxf(c[r], 0.0f);
        int g = batch[row];
        atomicAdd(&pool_sum[g * N_CLASSES + n], v);
    }
}

__global__ void count_nodes(const int* __restrict__ batch,
                            float* __restrict__ pool_cnt) {
    int i = blockIdx.x * blockDim.x + threadIdx.x;
    if (i < N_NODES) atomicAdd(pool_cnt + batch[i], 1.0f);
}

__global__ void finalize_logsoftmax(const float* __restrict__ pool_sum,
                                    const float* __restrict__ pool_cnt,
                                    float* __restrict__ out) {
    int g = threadIdx.x;
    if (g >= N_GRAPHS) return;
    float inv = 1.0f / fmaxf(pool_cnt[g], 1.0f);
    float v[N_CLASSES];
    float mx = -3.0e38f;
    #pragma unroll
    for (int c = 0; c < N_CLASSES; ++c) {
        v[c] = pool_sum[g * N_CLASSES + c] * inv;
        mx = fmaxf(mx, v[c]);
    }
    float s = 0.0f;
    #pragma unroll
    for (int c = 0; c < N_CLASSES; ++c) s += __expf(v[c] - mx);
    float l = __logf(s);
    #pragma unroll
    for (int c = 0; c < N_CLASSES; ++c) out[g * N_CLASSES + c] = v[c] - mx - l;
}

// ---------------------------------------------------------------------------
extern "C" void kernel_launch(void* const* d_in, const int* in_sizes, int n_in,
                              void* d_out, int out_size, void* d_ws, size_t ws_size,
                              hipStream_t stream) {
    const float* x    = (const float*)d_in[0];
    const int*   ei   = (const int*)  d_in[1];
    const int*   bat  = (const int*)  d_in[2];
    const float* W1_l = (const float*)d_in[3];
    const float* W1_r = (const float*)d_in[4];
    const float* b1   = (const float*)d_in[5];
    const float* W2_l = (const float*)d_in[6];
    const float* W2_r = (const float*)d_in[7];
    const float* b2   = (const float*)d_in[8];
    float* out = (float*)d_out;

    // Workspace layout (float region, then bf16 region; bf16 start is 32B
    // aligned: 32,051,088 floats = 128,204,352 bytes, divisible by 32).
    float* ws   = (float*)d_ws;
    float* msg1 = ws;                                        // N*F_IN
    float* cnt  = msg1 + (size_t)N_NODES * F_IN;             // N
    float* h1   = cnt  + N_NODES;                            // N*H_DIM
    float* msg2 = h1   + (size_t)N_NODES * H_DIM;            // N*H_DIM
    float* psum = msg2 + (size_t)N_NODES * H_DIM;            // G*C
    float* pcnt = psum + N_GRAPHS * N_CLASSES;               // G
    __bf16* w1l_bf = (__bf16*)(pcnt + N_GRAPHS);             // F_IN*H_DIM
    __bf16* w1r_bf = w1l_bf + (size_t)F_IN * H_DIM;
    __bf16* w2l_bf = w1r_bf + (size_t)F_IN * H_DIM;          // H_DIM*N_CLASSES
    __bf16* w2r_bf = w2l_bf + (size_t)H_DIM * N_CLASSES;

    // Zero accumulators (msg1+cnt contiguous; msg2+psum+pcnt contiguous)
    hipMemsetAsync(msg1, 0, ((size_t)N_NODES * F_IN + N_NODES) * sizeof(float), stream);
    hipMemsetAsync(msg2, 0, ((size_t)N_NODES * H_DIM + N_GRAPHS * N_CLASSES + N_GRAPHS) * sizeof(float), stream);

    // One-shot weight re-tiling to bf16 WMMA-B layout.
    convert_w<<<(F_IN * H_DIM + 255) / 256, 256, 0, stream>>>(W1_l, w1l_bf, F_IN, H_DIM);
    convert_w<<<(F_IN * H_DIM + 255) / 256, 256, 0, stream>>>(W1_r, w1r_bf, F_IN, H_DIM);
    convert_w<<<(H_DIM * N_CLASSES + 255) / 256, 256, 0, stream>>>(W2_l, w2l_bf, H_DIM, N_CLASSES);
    convert_w<<<(H_DIM * N_CLASSES + 255) / 256, 256, 0, stream>>>(W2_r, w2r_bf, H_DIM, N_CLASSES);

    // Layer 1 aggregate + count
    scatter_mean1<<<N_EDGES / 8, 256, 0, stream>>>(x, ei, msg1, cnt);
    count_nodes<<<(N_NODES + 255) / 256, 256, 0, stream>>>(bat, pcnt);

    // Layer 1 GEMM (WMMA bf16, LDS-staged A)
    gemm1_wmma<<<N_NODES / 16, 512, 0, stream>>>(msg1, cnt, x, w1l_bf, w1r_bf, b1, h1);

    // Layer 2 aggregate
    scatter_mean2<<<N_EDGES / 8, 256, 0, stream>>>(h1, ei, msg2);

    // Layer 2 GEMM + fused relu + pool
    gemm2_wmma_pool<<<(N_NODES / 16 + 7) / 8, 256, 0, stream>>>(
        msg2, cnt, h1, w2l_bf, w2r_bf, b2, bat, psum);

    // Mean pool + log_softmax
    finalize_logsoftmax<<<1, 64, 0, stream>>>(psum, pcnt, out);
}
